// ResDeformConvBlock_27917287424179
// MI455X (gfx1250) — matrix operand order repaired and checked by reading, not compiled, not run on hardware
//
#include <hip/hip_runtime.h>
#include <math.h>

// ---------------- problem constants ----------------
#define Bn   4
#define Cc   128
#define Hh   112
#define Ww   112
#define Pp   (Hh * Ww)      // 12544 spatial positions
#define K2   9
#define CK   (Cc * K2)      // 1152 contraction length, flattened k-major: ck = k*128 + c
#define NKC  (CK / 32)      // 36 K-chunks of 32
#define NT   64             // positions per workgroup (4 WMMA N-subtiles)
#define NPT  (Pp / NT)      // 196 position tiles
#define OFFC (2 * K2)       // 18 offset channels
#define NTOT ((size_t)Bn * Cc * Pp)

typedef __attribute__((ext_vector_type(16))) __bf16 v16bf;
typedef __attribute__((ext_vector_type(8)))  float  v8f;

__device__ __forceinline__ unsigned short f2bf(float f) {
  unsigned int u = __float_as_uint(f);
  u += 0x7FFFu + ((u >> 16) & 1u);   // round to nearest even
  return (unsigned short)(u >> 16);
}

// ---------------- weight prepack: f32 -> bf16 in WMMA A-fragment lane order ----
// wpack[s], s = ((blk*8 + mtile)*36 + kchunk)*512 + lane*16 + elem
// Contraction index is k-major: ck = k*128 + c  ->  w element [o][c][k] = w[o*1152 + c*9 + k].
// A k order used here (contiguous, same as B): lanes 0-15 M=lane, elements K=0..15;
//                        lanes 16-31 M=lane-16, elements K=16..31.
__global__ void prepack_w(const float* __restrict__ w1, const float* __restrict__ w2,
                          unsigned short* __restrict__ wpack) {
  int s = blockIdx.x * blockDim.x + threadIdx.x;
  if (s >= 2 * 8 * NKC * 512) return;
  const float* w = (s < 8 * NKC * 512) ? w1 : w2;
  int rem  = s % (8 * NKC * 512);
  int i    = rem & 15;
  int lane = (rem >> 4) & 31;
  int kc   = (rem >> 9) % NKC;
  int mt   = rem / (512 * NKC);
  int M = lane & 15, half = lane >> 4, v = i >> 1, pos = i & 1;
  int K = 16 * half + 2 * v + pos;
  int o  = mt * 16 + M;
  int ck = kc * 32 + K;          // k-major flatten
  int k  = ck >> 7;              // tap 0..8
  int c  = ck & 127;             // channel
  wpack[s] = f2bf(w[o * CK + c * K2 + k]);
}

// ---------------- offset conv: 3x3, 128 -> 18 channels (direct, L2-resident) ----
__global__ void offset_conv(const float* __restrict__ x, const float* __restrict__ wo,
                            const float* __restrict__ bo, float* __restrict__ off) {
  int idx = blockIdx.x * blockDim.x + threadIdx.x;
  if (idx >= Bn * OFFC * Pp) return;
  int p  = idx % Pp;
  int oc = (idx / Pp) % OFFC;
  int b  = idx / (Pp * OFFC);
  int oy = p / Ww, ox = p % Ww;
  int tIdx[9];
#pragma unroll
  for (int t = 0; t < 9; ++t) {
    int yy = oy - 1 + t / 3, xx = ox - 1 + t % 3;
    tIdx[t] = (yy >= 0 && yy < Hh && xx >= 0 && xx < Ww) ? yy * Ww + xx : -1;
  }
  float acc = bo[oc];
  const float* wb = wo + (size_t)oc * CK;
  for (int c = 0; c < Cc; ++c) {
    const float* xb = x + ((size_t)b * Cc + c) * Pp;
    const float* wc = wb + c * 9;
#pragma unroll
    for (int t = 0; t < 9; ++t)
      if (tIdx[t] >= 0) acc = fmaf(wc[t], xb[tIdx[t]], acc);
  }
  off[idx] = acc;
}

// ---------------- deformable conv as bf16 WMMA GEMM ----------------
// One workgroup: batch b, 64-position tile. 8 waves = 8 M-tiles (128 outputs),
// each wave does 4 N-subtile WMMAs per K-chunk sharing one A fragment.
// k-major chunks: tap k = kc>>2 is uniform per chunk (scalar tap-table base).
__global__ __launch_bounds__(256)
void deform_wmma(const float* __restrict__ x, const float* __restrict__ off,
                 const unsigned short* __restrict__ wpack, int wblk,
                 float* __restrict__ yraw) {
  __shared__ int   sIdx[K2 * NT * 4];   // per (tap k, pos p): 4 corner indices
  __shared__ float sWgt[K2 * NT * 4];   // 4 bilinear weights (zeroed when OOB)
  __shared__ __align__(32) unsigned short smp[NT * 48];  // [p][ck_local], padded rows

  const int tid = threadIdx.x;
  const int b   = blockIdx.x / NPT;
  const int pt  = blockIdx.x % NPT;

  // ---- precompute bilinear taps (independent of channel) ----
  for (int e = tid; e < K2 * NT; e += 256) {
    int k = e / NT, p = e % NT;
    int pg = pt * NT + p;
    int oy = pg / Ww, ox = pg % Ww;
    float dy = off[((size_t)b * OFFC + 2 * k)     * Pp + pg];
    float dx = off[((size_t)b * OFFC + 2 * k + 1) * Pp + pg];
    float py = (float)(oy - 1 + k / 3) + dy;
    float px = (float)(ox - 1 + k % 3) + dx;
    float fy = floorf(py), fx = floorf(px);
    float wy = py - fy, wx = px - fx;
    int y0 = (int)fy, x0 = (int)fx;
    float cw[4] = {(1.f - wy) * (1.f - wx), (1.f - wy) * wx,
                   wy * (1.f - wx),         wy * wx};
    int yy[4] = {y0, y0, y0 + 1, y0 + 1};
    int xx[4] = {x0, x0 + 1, x0, x0 + 1};
#pragma unroll
    for (int j = 0; j < 4; ++j) {
      bool ok = (yy[j] >= 0 && yy[j] < Hh && xx[j] >= 0 && xx[j] < Ww);
      sIdx[e * 4 + j] = ok ? yy[j] * Ww + xx[j] : 0;
      sWgt[e * 4 + j] = ok ? cw[j] : 0.f;
    }
  }
  __syncthreads();

  const int wv    = tid >> 5;         // wave id == M tile
  const int lane  = tid & 31;
  const int ckl   = tid >> 3;         // sampling: ck_local 0..31
  const int pbase = (tid & 7) * 8;    // sampling: 8 consecutive p
  v8f acc0 = {}, acc1 = {}, acc2 = {}, acc3 = {};
  const v16bf* wp = reinterpret_cast<const v16bf*>(wpack);
  const int abase = (wblk * 8 + wv) * NKC;

  for (int kc = 0; kc < NKC; ++kc) {
    // ---- sample 32ck x 64p tile into LDS (bf16); 4-corner gathers hit L2 ----
    {
      int k = kc >> 2;                       // uniform tap for whole chunk
      int c = ((kc & 3) << 5) + ckl;         // channel
      const float* xc = x + ((size_t)b * Cc + c) * Pp;
      const int ebase = k * NT * 4;
#pragma unroll
      for (int j = 0; j < 8; ++j) {
        int p  = pbase + j;
        int e4 = ebase + p * 4;
        float v = sWgt[e4]     * xc[sIdx[e4]]
                + sWgt[e4 + 1] * xc[sIdx[e4 + 1]]
                + sWgt[e4 + 2] * xc[sIdx[e4 + 2]]
                + sWgt[e4 + 3] * xc[sIdx[e4 + 3]];
        smp[p * 48 + ckl] = f2bf(v);
      }
    }
    __syncthreads();

    // ---- A fragment: one contiguous 32B load per lane (prepacked layout) ----
    v16bf a = wp[(abase + kc) * 32 + lane];
    if (kc + 1 < NKC)  // global_prefetch of next A chunk
      __builtin_prefetch(&wpack[(size_t)((abase + kc + 1) * 32 + lane) * 16], 0, 0);

    // ---- B fragments from LDS: lane n holds K=0..15 (lanes 0-15) / 16..31 ----
    int prow = lane & 15;
    int ck0  = (lane < 16) ? 0 : 16;
    v16bf b0 = *reinterpret_cast<const v16bf*>(&smp[(prow)      * 48 + ck0]);
    v16bf b1 = *reinterpret_cast<const v16bf*>(&smp[(16 + prow) * 48 + ck0]);
    v16bf b2 = *reinterpret_cast<const v16bf*>(&smp[(32 + prow) * 48 + ck0]);
    v16bf b3 = *reinterpret_cast<const v16bf*>(&smp[(48 + prow) * 48 + ck0]);

    acc0 = __builtin_amdgcn_wmma_f32_16x16x32_bf16(false, a, false, b0,
                                                   (short)0, acc0, false, false);
    acc1 = __builtin_amdgcn_wmma_f32_16x16x32_bf16(false, a, false, b1,
                                                   (short)0, acc1, false, false);
    acc2 = __builtin_amdgcn_wmma_f32_16x16x32_bf16(false, a, false, b2,
                                                   (short)0, acc2, false, false);
    acc3 = __builtin_amdgcn_wmma_f32_16x16x32_bf16(false, a, false, b3,
                                                   (short)0, acc3, false, false);
    __syncthreads();
  }

  // ---- store D (16x16 f32 layout: VGPR r -> M = r + (lane<16?0:8), N = lane%16)
  int ncol = lane & 15;
  int mofs = (lane < 16) ? 0 : 8;
#pragma unroll
  for (int r = 0; r < 8; ++r) {
    int o = wv * 16 + r + mofs;
    size_t base = ((size_t)b * Cc + o) * Pp + (size_t)pt * NT;
    yraw[base + ncol]      = acc0[r];
    yraw[base + 16 + ncol] = acc1[r];
    yraw[base + 32 + ncol] = acc2[r];
    yraw[base + 48 + ncol] = acc3[r];
  }
}

// ---------------- per-channel batch-norm statistics ----------------
__global__ void bn_stats(const float* __restrict__ y, float* __restrict__ mean,
                         float* __restrict__ var) {
  __shared__ float rs[256], rq[256];
  int ch = blockIdx.x, tid = threadIdx.x;
  float s = 0.f, q = 0.f;
  for (int i = tid; i < Bn * Pp; i += 256) {
    int b = i / Pp, p = i % Pp;
    float v = y[((size_t)b * Cc + ch) * Pp + p];
    s += v; q += v * v;
  }
  rs[tid] = s; rq[tid] = q;
  __syncthreads();
  for (int st = 128; st > 0; st >>= 1) {
    if (tid < st) { rs[tid] += rs[tid + st]; rq[tid] += rq[tid + st]; }
    __syncthreads();
  }
  if (tid == 0) {
    float inv = 1.f / (float)(Bn * Pp);
    float m = rs[0] * inv;
    mean[ch] = m;
    var[ch]  = rq[0] * inv - m * m;
  }
}

// ---------------- BN + exact GELU (+ optional residual) ----------------
__global__ void bn_gelu_apply(const float* __restrict__ y, const float* __restrict__ mean,
                              const float* __restrict__ var, const float* __restrict__ gamma,
                              const float* __restrict__ beta, const float* __restrict__ resid,
                              float* __restrict__ out) {
  size_t idx = (size_t)blockIdx.x * blockDim.x + threadIdx.x;
  if (idx >= NTOT) return;
  int o = (int)((idx / Pp) % Cc);
  float xn = (y[idx] - mean[o]) * rsqrtf(var[o] + 1e-5f) * gamma[o] + beta[o];
  float g  = 0.5f * xn * (1.f + erff(xn * 0.70710678118654752440f));
  if (resid) g += resid[idx];
  out[idx] = g;
}

// ---------------- host-side launcher ----------------
extern "C" void kernel_launch(void* const* d_in, const int* in_sizes, int n_in,
                              void* d_out, int out_size, void* d_ws, size_t ws_size,
                              hipStream_t stream) {
  const float* x     = (const float*)d_in[0];
  const float* woff1 = (const float*)d_in[1];
  const float* boff1 = (const float*)d_in[2];
  const float* wdef1 = (const float*)d_in[3];
  const float* gam1  = (const float*)d_in[4];
  const float* bet1  = (const float*)d_in[5];
  const float* woff2 = (const float*)d_in[6];
  const float* boff2 = (const float*)d_in[7];
  const float* wdef2 = (const float*)d_in[8];
  const float* gam2  = (const float*)d_in[9];
  const float* bet2  = (const float*)d_in[10];
  float* out = (float*)d_out;

  // workspace carve-up (all 256B-aligned)
  char* ws = (char*)d_ws;
  const size_t offBytes = (size_t)Bn * OFFC * Pp * 4;    // 3.6 MB
  const size_t yBytes   = NTOT * 4;                      // 25.7 MB
  const size_t wpBytes  = (size_t)2 * 8 * NKC * 512 * 2; // bf16 packed weights
  float*          off_buf = (float*)ws;
  float*          y_raw   = (float*)(ws + offBytes);
  float*          x1      = (float*)(ws + offBytes + yBytes);
  unsigned short* wpack   = (unsigned short*)(ws + offBytes + 2 * yBytes);
  float*          mean    = (float*)(ws + offBytes + 2 * yBytes + ((wpBytes + 255) & ~(size_t)255));
  float*          var     = mean + 128;

  const int nPackThreads = 2 * 8 * NKC * 512;
  prepack_w<<<(nPackThreads + 255) / 256, 256, 0, stream>>>(wdef1, wdef2, wpack);

  // ---- block 1 ----
  offset_conv<<<(Bn * OFFC * Pp + 255) / 256, 256, 0, stream>>>(x, woff1, boff1, off_buf);
  deform_wmma<<<Bn * NPT, 256, 0, stream>>>(x, off_buf, wpack, 0, y_raw);
  bn_stats<<<Cc, 256, 0, stream>>>(y_raw, mean, var);
  bn_gelu_apply<<<(unsigned)((NTOT + 255) / 256), 256, 0, stream>>>(
      y_raw, mean, var, gam1, bet1, nullptr, x1);

  // ---- block 2 + residual ----
  offset_conv<<<(Bn * OFFC * Pp + 255) / 256, 256, 0, stream>>>(x1, woff2, boff2, off_buf);
  deform_wmma<<<Bn * NPT, 256, 0, stream>>>(x1, off_buf, wpack, 1, y_raw);
  bn_stats<<<Cc, 256, 0, stream>>>(y_raw, mean, var);
  bn_gelu_apply<<<(unsigned)((NTOT + 255) / 256), 256, 0, stream>>>(
      y_raw, mean, var, gam2, bet2, x, out);
}
